// MyTransformer_81312320848268
// MI455X (gfx1250) — matrix operand order repaired and checked by reading, not compiled, not run on hardware
//
#include <hip/hip_runtime.h>

#define NS 1024   // NUM_SHOT
#define DM 512    // D_MODEL
#define FF 2048   // D_FF
#define NH 8      // heads
#define HD 64     // head dim
#define NL 6      // layers

typedef __attribute__((ext_vector_type(16))) _Float16 v16h;
typedef __attribute__((ext_vector_type(8)))  _Float16 v8h;
typedef __attribute__((ext_vector_type(4)))  _Float16 v4h;
typedef __attribute__((ext_vector_type(8)))  float    v8f;
typedef __attribute__((ext_vector_type(4)))  float    f32x4;

__device__ __forceinline__ v16h cat8(v8h lo, v8h hi) {
  v16h r;
#pragma unroll
  for (int i = 0; i < 8; ++i) { r[i] = lo[i]; r[i + 8] = hi[i]; }
  return r;
}

// C[m,n] = RELU?( alpha * sum_k A[m,k]*Bm[k,n] + bias[n] )
// TRANSB=1: Bm[k,n] = B[n*ldb+k]   (row-major [N,K] weights -> X @ W^T)
// TRANSB=0: Bm[k,n] = B[k*ldb+n]   (plain NN, used for P @ V)
// grid.z batches (per-head views) via aBatch/bBatch/cBatch element strides.
// 64x64 block tile, 4 wave32s, each wave = 2x2 v_wmma_f32_16x16x32_f16.
// Double-buffered LDS; next tile prefetched into registers via b128 loads
// while the current tile's WMMAs run; one barrier per K-step.
// Requires: lda/ldb/ldc multiples of 4, 16B-aligned base pointers, K%32==0,
// M%64==0, N%64==0 (true for every GEMM in this model).
template <int TRANSB, int RELU, int HASBIAS>
__global__ __launch_bounds__(128) void k_gemm_wmma(
    const float* __restrict__ A, const float* __restrict__ B,
    const float* __restrict__ bias, float* __restrict__ C,
    int K, int lda, int ldb, int ldc,
    long aBatch, long bBatch, long cBatch, float alpha)
{
  __shared__ alignas(16) _Float16 Af[2][64][40];   // [buf][m][k], padded pitch
  __shared__ alignas(16) _Float16 Bf[2][64][40];   // [buf][n][k], padded pitch

  const int tid  = threadIdx.x;
  const int wave = tid >> 5;
  const int lane = tid & 31;
  const int half = lane >> 4;     // 16-lane half of the wave
  const int l16  = lane & 15;
  const int bm = blockIdx.y * 64;
  const int bn = blockIdx.x * 64;
  const int wm = (wave >> 1) * 32;
  const int wn = (wave & 1) * 32;

  A += (long)blockIdx.z * aBatch;
  B += (long)blockIdx.z * bBatch;
  C += (long)blockIdx.z * cBatch;

  f32x4 ra[4], rb[4];

  // A tile 64x32: float4 index f -> row f>>3, col4 (f&7)*4  (coalesced on k)
  auto loadA = [&](int k0) {
#pragma unroll
    for (int i = 0; i < 4; ++i) {
      int f = i * 128 + tid;
      ra[i] = *(const f32x4*)&A[(long)(bm + (f >> 3)) * lda + k0 + ((f & 7) << 2)];
    }
  };
  // B tile as [n][k]: TRANSB=1 mirrors A; TRANSB=0 reads [k][n] coalesced on n
  auto loadB = [&](int k0) {
    if (TRANSB) {
#pragma unroll
      for (int i = 0; i < 4; ++i) {
        int f = i * 128 + tid;
        rb[i] = *(const f32x4*)&B[(long)(bn + (f >> 3)) * ldb + k0 + ((f & 7) << 2)];
      }
    } else {
#pragma unroll
      for (int i = 0; i < 4; ++i) {
        int f = i * 128 + tid;     // f>>4 = k row (0..31), (f&15)*4 = n
        rb[i] = *(const f32x4*)&B[(long)(k0 + (f >> 4)) * ldb + bn + ((f & 15) << 2)];
      }
    }
  };

  v8f acc[2][2] = {};
  const int nk = K >> 5;

  loadA(0);
  loadB(0);
  int cur = 0;

  for (int t = 0; t < nk; ++t) {
    // stage registers -> f16 LDS[cur]  (v_cvt_pk_f16_f32 + ds_store_b64)
#pragma unroll
    for (int i = 0; i < 4; ++i) {
      int f = i * 128 + tid;
      v4h h;
#pragma unroll
      for (int j = 0; j < 4; ++j) h[j] = (_Float16)ra[i][j];
      *(v4h*)&Af[cur][f >> 3][(f & 7) << 2] = h;
    }
    if (TRANSB) {
#pragma unroll
      for (int i = 0; i < 4; ++i) {
        int f = i * 128 + tid;
        v4h h;
#pragma unroll
        for (int j = 0; j < 4; ++j) h[j] = (_Float16)rb[i][j];
        *(v4h*)&Bf[cur][f >> 3][(f & 7) << 2] = h;
      }
    } else {
#pragma unroll
      for (int i = 0; i < 4; ++i) {
        int f = i * 128 + tid;
        int kk = f >> 4, n4 = (f & 15) << 2;
#pragma unroll
        for (int j = 0; j < 4; ++j) Bf[cur][n4 + j][kk] = (_Float16)rb[i][j];
      }
    }
    __syncthreads();

    // prefetch next K-tile into registers while WMMAs run
    if (t + 1 < nk) { loadA((t + 1) << 5); loadB((t + 1) << 5); }

    // Fragment assembly per ISA 7.12.2 (16-bit A 16x32 / B 32x16, wave32):
    //   A lane(l16,half): K = 8*half..+7 in v0..3, 16+8*half..+7 in v4..7
    //   B lane(l16,half): K = 8*half..+7 in v0..3, 16+8*half..+7 in v4..7
    v16h afr[2], bfr[2];
#pragma unroll
    for (int q = 0; q < 2; ++q) {
      const _Float16* ap = &Af[cur][wm + q * 16 + l16][0];
      afr[q] = cat8(*(const v8h*)(ap + (half << 3)),
                    *(const v8h*)(ap + 16 + (half << 3)));
      const _Float16* bp = &Bf[cur][wn + q * 16 + l16][0];
      bfr[q] = cat8(*(const v8h*)(bp + (half << 3)),
                    *(const v8h*)(bp + 16 + (half << 3)));
    }
#pragma unroll
    for (int i = 0; i < 2; ++i)
#pragma unroll
      for (int j = 0; j < 2; ++j)
        acc[i][j] = __builtin_amdgcn_wmma_f32_16x16x32_f16(
            false, afr[i], false, bfr[j], (short)0, acc[i][j], false, false);

    cur ^= 1;
  }

  // D layout: vgpr r -> row = r + 8*half (within 16), col = l16
#pragma unroll
  for (int i = 0; i < 2; ++i) {
#pragma unroll
    for (int j = 0; j < 2; ++j) {
      int col = bn + wn + j * 16 + l16;
      float bv = HASBIAS ? bias[col] : 0.0f;
#pragma unroll
      for (int r = 0; r < 8; ++r) {
        int row = bm + wm + i * 16 + (half << 3) + r;
        float v = alpha * acc[i][j][r] + bv;
        if (RELU) v = v > 0.0f ? v : 0.0f;
        C[(long)row * ldc + col] = v;
      }
    }
  }
}

__global__ __launch_bounds__(256) void k_softmax(float* __restrict__ S, int cols) {
  float* p = S + (long)blockIdx.x * cols;
  const int tid = threadIdx.x;
  __shared__ float red[256];
  float m = -3.4e38f;
  for (int c = tid; c < cols; c += 256) m = fmaxf(m, p[c]);
  red[tid] = m; __syncthreads();
  for (int s = 128; s > 0; s >>= 1) {
    if (tid < s) red[tid] = fmaxf(red[tid], red[tid + s]);
    __syncthreads();
  }
  m = red[0]; __syncthreads();
  float sum = 0.0f;
  for (int c = tid; c < cols; c += 256) {
    float e = __expf(p[c] - m); p[c] = e; sum += e;
  }
  red[tid] = sum; __syncthreads();
  for (int s = 128; s > 0; s >>= 1) {
    if (tid < s) red[tid] += red[tid + s];
    __syncthreads();
  }
  float inv = 1.0f / red[0];
  for (int c = tid; c < cols; c += 256) p[c] *= inv;
}

// out = LN(x + res) * s + b ; res may be null. Row width fixed to DM=512.
__global__ __launch_bounds__(256) void k_add_ln(
    const float* __restrict__ x, const float* __restrict__ res,
    const float* __restrict__ s, const float* __restrict__ b,
    float* __restrict__ out)
{
  const int row = blockIdx.x, tid = threadIdx.x;
  const float* xp = x + (long)row * DM;
  float v0 = xp[tid], v1 = xp[tid + 256];
  if (res) {
    const float* rp = res + (long)row * DM;
    v0 += rp[tid]; v1 += rp[tid + 256];
  }
  __shared__ float rs[256], rq[256];
  rs[tid] = v0 + v1; rq[tid] = v0 * v0 + v1 * v1;
  __syncthreads();
  for (int st = 128; st > 0; st >>= 1) {
    if (tid < st) { rs[tid] += rs[tid + st]; rq[tid] += rq[tid + st]; }
    __syncthreads();
  }
  float mean = rs[0] * (1.0f / DM);
  float var  = rq[0] * (1.0f / DM) - mean * mean;
  float inv  = rsqrtf(var + 1e-5f);
  float* op = out + (long)row * DM;
  op[tid]       = (v0 - mean) * inv * s[tid]       + b[tid];
  op[tid + 256] = (v1 - mean) * inv * s[tid + 256] + b[tid + 256];
}

__global__ __launch_bounds__(256) void k_rownorm(
    const float* __restrict__ x, float* __restrict__ o)
{
  const int row = blockIdx.x, tid = threadIdx.x;
  const float* xp = x + (long)row * DM;
  float v0 = xp[tid], v1 = xp[tid + 256];
  __shared__ float rq[256];
  rq[tid] = v0 * v0 + v1 * v1; __syncthreads();
  for (int st = 128; st > 0; st >>= 1) {
    if (tid < st) rq[tid] += rq[tid + st];
    __syncthreads();
  }
  float inv = rsqrtf(rq[0]);
  float* op = o + (long)row * DM;
  op[tid] = v0 * inv; op[tid + 256] = v1 * inv;
}

// Host-side dispatch over the four configurations actually used.
static void gemm(hipStream_t st, const float* A, const float* B,
                 const float* bias, float* C,
                 int M, int N, int K, int lda, int ldb, int ldc,
                 long aB, long bB, long cB, float alpha,
                 int transB, int relu, int Z)
{
  dim3 grid(N / 64, M / 64, Z), blk(128, 1, 1);
  if (transB) {
    if (relu)
      k_gemm_wmma<1, 1, 1><<<grid, blk, 0, st>>>(A, B, bias, C, K, lda, ldb, ldc,
                                                 aB, bB, cB, alpha);
    else if (bias)
      k_gemm_wmma<1, 0, 1><<<grid, blk, 0, st>>>(A, B, bias, C, K, lda, ldb, ldc,
                                                 aB, bB, cB, alpha);
    else
      k_gemm_wmma<1, 0, 0><<<grid, blk, 0, st>>>(A, B, bias, C, K, lda, ldb, ldc,
                                                 aB, bB, cB, alpha);
  } else {
    k_gemm_wmma<0, 0, 0><<<grid, blk, 0, st>>>(A, B, bias, C, K, lda, ldb, ldc,
                                               aB, bB, cB, alpha);
  }
}

extern "C" void kernel_launch(void* const* d_in, const int* in_sizes, int n_in,
                              void* d_out, int out_size, void* d_ws, size_t ws_size,
                              hipStream_t stream)
{
  (void)in_sizes; (void)n_in; (void)out_size; (void)ws_size;
  const float* x          = (const float*)d_in[0];
  const float* enc_qkv_w  = (const float*)d_in[1];
  const float* enc_qkv_b  = (const float*)d_in[2];
  const float* enc_out_w  = (const float*)d_in[3];
  const float* enc_out_b  = (const float*)d_in[4];
  const float* enc_ff1_w  = (const float*)d_in[5];
  const float* enc_ff1_b  = (const float*)d_in[6];
  const float* enc_ff2_w  = (const float*)d_in[7];
  const float* enc_ff2_b  = (const float*)d_in[8];
  const float* enc_ln1_s  = (const float*)d_in[9];
  const float* enc_ln1_b  = (const float*)d_in[10];
  const float* enc_ln2_s  = (const float*)d_in[11];
  const float* enc_ln2_b  = (const float*)d_in[12];
  const float* dsa_qkv_w  = (const float*)d_in[13];
  const float* dsa_qkv_b  = (const float*)d_in[14];
  const float* dsa_out_w  = (const float*)d_in[15];
  const float* dsa_out_b  = (const float*)d_in[16];
  const float* dca_qkv_w  = (const float*)d_in[17];
  const float* dca_qkv_b  = (const float*)d_in[18];
  const float* dca_out_w  = (const float*)d_in[19];
  const float* dca_out_b  = (const float*)d_in[20];
  const float* dec_ff1_w  = (const float*)d_in[21];
  const float* dec_ff1_b  = (const float*)d_in[22];
  const float* dec_ff2_w  = (const float*)d_in[23];
  const float* dec_ff2_b  = (const float*)d_in[24];
  const float* dec_ln1_s  = (const float*)d_in[25];
  const float* dec_ln1_b  = (const float*)d_in[26];
  const float* dec_ln2_s  = (const float*)d_in[27];
  const float* dec_ln2_b  = (const float*)d_in[28];
  const float* dec_ln3_s  = (const float*)d_in[29];
  const float* dec_ln3_b  = (const float*)d_in[30];
  const float* enc_norm_s = (const float*)d_in[31];
  const float* enc_norm_b = (const float*)d_in[32];
  const float* dec_norm_s = (const float*)d_in[33];
  const float* dec_norm_b = (const float*)d_in[34];

  const long SZX = (long)NS * DM;          // 1024*512
  float* ws   = (float*)d_ws;
  float* mem  = ws;                        // 1024x512
  float* y    = mem + SZX;                 // 1024x512
  float* qkv  = y + SZX;                   // 1024x1536 (also Q | KV for cross)
  float* attn = qkv + (long)NS * 3 * DM;   // 1024x512
  float* t0   = attn + SZX;                // 1024x512
  float* ff   = t0 + SZX;                  // 1024x2048
  float* sc   = ff + (long)NS * FF;        // 8x1024x1024

  float* outG    = (float*)d_out;          // final_result 1024x1024
  float* attnOut = outG + (long)NS * NS;   // attention_out 1024x512

  hipMemcpyAsync(mem, x, SZX * sizeof(float), hipMemcpyDeviceToDevice, stream);
  hipMemcpyAsync(y,   x, SZX * sizeof(float), hipMemcpyDeviceToDevice, stream);

  const float iscale = 0.125f;  // 1/sqrt(64)

  // ---------------- encoder ----------------
  for (int i = 0; i < NL; ++i) {
    const float* qw  = enc_qkv_w + (long)i * 3 * DM * DM;
    const float* qb  = enc_qkv_b + (long)i * 3 * DM;
    // fused QKV projection: [1024,512] @ [1536,512]^T
    gemm(stream, mem, qw, qb, qkv, NS, 3 * DM, DM, DM, DM, 3 * DM,
         0, 0, 0, 1.0f, 1, 0, 1);
    // per-head scores = (Q_h K_h^T)/8, batched over z=8 heads
    gemm(stream, qkv, qkv + DM, nullptr, sc, NS, NS, HD, 3 * DM, 3 * DM, NS,
         HD, HD, (long)NS * NS, iscale, 1, 0, NH);
    k_softmax<<<NH * NS, 256, 0, stream>>>(sc, NS);
    // O_h = P_h @ V_h (NN)
    gemm(stream, sc, qkv + 2 * DM, nullptr, attn, NS, HD, NS, NS, 3 * DM, DM,
         (long)NS * NS, HD, HD, 1.0f, 0, 0, NH);
    gemm(stream, attn, enc_out_w + (long)i * DM * DM, enc_out_b + (long)i * DM,
         t0, NS, DM, DM, DM, DM, DM, 0, 0, 0, 1.0f, 1, 0, 1);
    k_add_ln<<<NS, 256, 0, stream>>>(mem, t0,
        enc_ln1_s + (long)i * DM, enc_ln1_b + (long)i * DM, mem);
    // FFN
    gemm(stream, mem, enc_ff1_w + (long)i * FF * DM, enc_ff1_b + (long)i * FF,
         ff, NS, FF, DM, DM, DM, FF, 0, 0, 0, 1.0f, 1, 1, 1);
    gemm(stream, ff, enc_ff2_w + (long)i * DM * FF, enc_ff2_b + (long)i * DM,
         t0, NS, DM, FF, FF, FF, DM, 0, 0, 0, 1.0f, 1, 0, 1);
    k_add_ln<<<NS, 256, 0, stream>>>(mem, t0,
        enc_ln2_s + (long)i * DM, enc_ln2_b + (long)i * DM, mem);
  }
  k_add_ln<<<NS, 256, 0, stream>>>(mem, nullptr, enc_norm_s, enc_norm_b, mem);

  // ---------------- decoder ----------------
  for (int i = 0; i < NL; ++i) {
    // self-attention
    const float* qw = dsa_qkv_w + (long)i * 3 * DM * DM;
    const float* qb = dsa_qkv_b + (long)i * 3 * DM;
    gemm(stream, y, qw, qb, qkv, NS, 3 * DM, DM, DM, DM, 3 * DM,
         0, 0, 0, 1.0f, 1, 0, 1);
    gemm(stream, qkv, qkv + DM, nullptr, sc, NS, NS, HD, 3 * DM, 3 * DM, NS,
         HD, HD, (long)NS * NS, iscale, 1, 0, NH);
    k_softmax<<<NH * NS, 256, 0, stream>>>(sc, NS);
    gemm(stream, sc, qkv + 2 * DM, nullptr, attn, NS, HD, NS, NS, 3 * DM, DM,
         (long)NS * NS, HD, HD, 1.0f, 0, 0, NH);
    gemm(stream, attn, dsa_out_w + (long)i * DM * DM, dsa_out_b + (long)i * DM,
         t0, NS, DM, DM, DM, DM, DM, 0, 0, 0, 1.0f, 1, 0, 1);
    k_add_ln<<<NS, 256, 0, stream>>>(y, t0,
        dec_ln1_s + (long)i * DM, dec_ln1_b + (long)i * DM, y);

    // cross-attention: Q from y, K/V from encoder memory
    const float* cw = dca_qkv_w + (long)i * 3 * DM * DM;
    const float* cb = dca_qkv_b + (long)i * 3 * DM;
    float* Q  = qkv;         // 1024x512
    float* KV = qkv + SZX;   // 1024x1024 (K | V)
    gemm(stream, y, cw, cb, Q, NS, DM, DM, DM, DM, DM, 0, 0, 0, 1.0f, 1, 0, 1);
    gemm(stream, mem, cw + (long)DM * DM, cb + DM, KV, NS, 2 * DM, DM,
         DM, DM, 2 * DM, 0, 0, 0, 1.0f, 1, 0, 1);
    gemm(stream, Q, KV, nullptr, sc, NS, NS, HD, DM, 2 * DM, NS,
         HD, HD, (long)NS * NS, iscale, 1, 0, NH);
    k_softmax<<<NH * NS, 256, 0, stream>>>(sc, NS);
    gemm(stream, sc, KV + DM, nullptr, attn, NS, HD, NS, NS, 2 * DM, DM,
         (long)NS * NS, HD, HD, 1.0f, 0, 0, NH);
    gemm(stream, attn, dca_out_w + (long)i * DM * DM, dca_out_b + (long)i * DM,
         t0, NS, DM, DM, DM, DM, DM, 0, 0, 0, 1.0f, 1, 0, 1);
    k_add_ln<<<NS, 256, 0, stream>>>(y, t0,
        dec_ln2_s + (long)i * DM, dec_ln2_b + (long)i * DM, y);

    // FFN
    gemm(stream, y, dec_ff1_w + (long)i * FF * DM, dec_ff1_b + (long)i * FF,
         ff, NS, FF, DM, DM, DM, FF, 0, 0, 0, 1.0f, 1, 1, 1);
    gemm(stream, ff, dec_ff2_w + (long)i * DM * FF, dec_ff2_b + (long)i * DM,
         t0, NS, DM, FF, FF, FF, DM, 0, 0, 0, 1.0f, 1, 0, 1);
    k_add_ln<<<NS, 256, 0, stream>>>(y, t0,
        dec_ln3_s + (long)i * DM, dec_ln3_b + (long)i * DM, y);
  }

  // attention_out = LN(y) written straight into its output slot
  k_add_ln<<<NS, 256, 0, stream>>>(y, nullptr, dec_norm_s, dec_norm_b, attnOut);
  // n = attention_out / ||row||;  final_result = n @ n^T (NT WMMA gemm)
  k_rownorm<<<NS, 256, 0, stream>>>(attnOut, attn);
  gemm(stream, attn, attn, nullptr, outG, NS, NS, DM, DM, DM, NS,
       0, 0, 0, 1.0f, 1, 0, 1);
}
